// LSTM_17643725652159
// MI455X (gfx1250) — matrix operand order repaired and checked by reading, not compiled, not run on hardware
//
#include <hip/hip_runtime.h>
#include <hip/hip_bf16.h>

// Problem dims
#define BB 32
#define TT 512
#define DD 1024
#define UU 1024
#define Z4 4096
#define NBLK 64            // UU / 16 blocks in the persistent scan kernel

typedef __attribute__((ext_vector_type(16))) __bf16 v16bf;
typedef __attribute__((ext_vector_type(8)))  float  v8f;

__device__ __forceinline__ __bf16 to_bf16(float f) {
  unsigned int u = __builtin_bit_cast(unsigned int, f);
  u += 0x7FFFu + ((u >> 16) & 1u);               // round-to-nearest-even
  unsigned short s = (unsigned short)(u >> 16);
  return __builtin_bit_cast(__bf16, s);
}

__device__ __forceinline__ float sigmoidf(float x) {
  return 1.0f / (1.0f + __expf(-x));
}

// ---------------------------------------------------------------------------
// Convert fp32 weights -> bf16 (kernel and recurrent_kernel, 4M elems each)
// ---------------------------------------------------------------------------
__global__ __launch_bounds__(256) void convert_weights_kernel(
    const float* __restrict__ W, const float* __restrict__ R,
    __bf16* __restrict__ Wb, __bf16* __restrict__ Rb) {
  size_t i = (size_t)blockIdx.x * 256 + threadIdx.x;
  if (i < (size_t)DD * Z4) {
    Wb[i] = to_bf16(W[i]);
    Rb[i] = to_bf16(R[i]);
  }
}

// ---------------------------------------------------------------------------
// LN1: xn = LayerNorm(x) ; writes fp32 xn (for residual) + bf16 xn (WMMA A)
// one block per (b,t) row of 1024
// ---------------------------------------------------------------------------
__global__ __launch_bounds__(256) void ln_in_kernel(
    const float* __restrict__ x, const float* __restrict__ gamma,
    const float* __restrict__ beta, float* __restrict__ xn,
    __bf16* __restrict__ xnb) {
  const int row = blockIdx.x;
  const int tid = threadIdx.x;
  const size_t base = (size_t)row * DD + tid * 4;
  float4 v = *(const float4*)(x + base);
  float s  = v.x + v.y + v.z + v.w;
  float ss = v.x * v.x + v.y * v.y + v.z * v.z + v.w * v.w;
#pragma unroll
  for (int o = 16; o > 0; o >>= 1) {
    s  += __shfl_down(s,  o, 32);
    ss += __shfl_down(ss, o, 32);
  }
  __shared__ float red[16];
  __shared__ float stat[2];
  const int lane = tid & 31, w = tid >> 5;
  if (lane == 0) { red[w] = s; red[8 + w] = ss; }
  __syncthreads();
  if (tid == 0) {
    float a = 0.f, q = 0.f;
#pragma unroll
    for (int i = 0; i < 8; ++i) { a += red[i]; q += red[8 + i]; }
    float mean = a * (1.0f / (float)DD);
    float var  = q * (1.0f / (float)DD) - mean * mean;
    stat[0] = mean;
    stat[1] = rsqrtf(var + 1e-3f);
  }
  __syncthreads();
  const float mean = stat[0], rstd = stat[1];
  float4 g = *(const float4*)(gamma + tid * 4);
  float4 b = *(const float4*)(beta + tid * 4);
  float4 y;
  y.x = (v.x - mean) * rstd * g.x + b.x;
  y.y = (v.y - mean) * rstd * g.y + b.y;
  y.z = (v.z - mean) * rstd * g.z + b.z;
  y.w = (v.w - mean) * rstd * g.w + b.w;
  *(float4*)(xn + base) = y;
  union { __bf16 e[4]; uint2 u; } pk;
  pk.e[0] = to_bf16(y.x); pk.e[1] = to_bf16(y.y);
  pk.e[2] = to_bf16(y.z); pk.e[3] = to_bf16(y.w);
  *(uint2*)((__bf16*)xnb + base) = pk.u;
}

// ---------------------------------------------------------------------------
// Persistent LSTM scan. 64 blocks x 256 threads (8 waves).
// Block j owns u-slice [16j,16j+16) of all 4 gates -> 64 z-columns.
// LDS: W,R column slices pre-swizzled into WMMA B-fragment order (256 KB),
// z staging (8 KB), cell state c (2 KB). Weights loaded once for all T steps.
// Wave w: gate g = w>>1, M-tile mi = w&1. K-loop 32 iters of 2 bf16 WMMAs.
// Step barrier: device-scope atomic counter, double-buffered h.
// ---------------------------------------------------------------------------
#define LDS_WF    0
#define LDS_RF    131072
#define LDS_Z     262144
#define LDS_C     270336
#define LDS_BYTES 272384

__global__ __launch_bounds__(256, 1) void lstm_scan_kernel(
    const __bf16* __restrict__ xnb, const __bf16* __restrict__ Wb,
    const __bf16* __restrict__ Rb, const float* __restrict__ bias,
    __bf16* __restrict__ hbuf, float* __restrict__ hlast,
    unsigned int* __restrict__ syncc) {
  extern __shared__ char smem[];
  __bf16* Wf  = (__bf16*)(smem + LDS_WF);   // [32 ktiles][4 gates][32 lanes][16]
  __bf16* Rf  = (__bf16*)(smem + LDS_RF);
  float*  zb  = (float*)(smem + LDS_Z);     // [32 rows][64 cols]
  float*  cst = (float*)(smem + LDS_C);     // [32 rows][16 u] cell state

  const int u0   = blockIdx.x * 16;
  const int tid  = threadIdx.x;
  const int lane = tid & 31;
  const int wave = tid >> 5;
  const int gi   = wave >> 1;               // gate 0..3 (i,f,g,o)
  const int mi   = wave & 1;                // M tile 0..1
  const int nn   = lane & 15;
  const int hi   = (lane & 16) ? 1 : 0;
  const int m    = mi * 16 + nn;            // batch row for A fragment

  // ---- one-time: pre-swizzle weight column slices into B-fragment order ----
  for (int i = tid; i < 65536; i += 256) {
    int j  = i & 15;                        // element within 32B fragment
    int ln = (i >> 4) & 31;                 // lane
    int g  = (i >> 9) & 3;                  // gate
    int kt = i >> 11;                       // k-tile
    int k  = kt * 32 + (j & 7) + ((ln & 16) ? 8 : 0) + ((j & 8) ? 16 : 0);   // B: lanes16-31 hold K+8, elems 8-15 K+16
    int col = g * 1024 + u0 + (ln & 15);           // B: n = lane&15
    Wf[i] = Wb[(size_t)k * Z4 + col];
    Rf[i] = Rb[(size_t)k * Z4 + col];
  }
  if (tid < 512) cst[tid] = 0.0f;
  __syncthreads();

  const float bv = bias[gi * 1024 + u0 + nn];      // C layout: n = lane&15
  const __bf16* xrow = xnb + (size_t)m * TT * DD + hi * 8;  // A: lanes16-31 K+8
  __bf16* hbuf0 = hbuf;
  __bf16* hbuf1 = hbuf + BB * UU;

  for (int t = 0; t < TT; ++t) {
    const __bf16* xr = xrow + (size_t)t * DD;
    const __bf16* hr = ((t & 1) ? hbuf1 : hbuf0) + m * UU + hi * 8;

    v8f acc = {bv, bv, bv, bv, bv, bv, bv, bv};
#pragma unroll 4
    for (int kt = 0; kt < 32; ++kt) {
      union { v16bf v; uint4 q[2]; } ax, ah;
      ax.q[0] = *(const uint4*)(xr + kt * 32);        // K 0..7  (+hi*8)
      ax.q[1] = *(const uint4*)(xr + kt * 32 + 16);   // K 16..23(+hi*8)
      ah.q[0] = *(const uint4*)(hr + kt * 32);
      ah.q[1] = *(const uint4*)(hr + kt * 32 + 16);
      const v16bf bw = *(const v16bf*)(Wf + (((kt * 4 + gi) * 32 + lane) << 4));
      const v16bf br = *(const v16bf*)(Rf + (((kt * 4 + gi) * 32 + lane) << 4));
      acc = __builtin_amdgcn_wmma_f32_16x16x32_bf16(false, ax.v, false, bw,
                                                    (short)0, acc, false, false);
      acc = __builtin_amdgcn_wmma_f32_16x16x32_bf16(false, ah.v, false, br,
                                                    (short)0, acc, false, false);
    }

    // scatter z tile into LDS: C layout VGPR r -> M=r (+8 for lanes 16-31)
    const int rbase = mi * 16 + (hi ? 8 : 0);
#pragma unroll
    for (int r = 0; r < 8; ++r)
      zb[(rbase + r) * 64 + gi * 16 + nn] = acc[r];
    __syncthreads();

    // fused gate nonlinearity + cell update (512 (row,u) pairs, 2 per thread)
    __bf16* hout = (t & 1) ? hbuf0 : hbuf1;
#pragma unroll
    for (int p = 0; p < 2; ++p) {
      int idx = tid + (p << 8);
      int row = idx >> 4;
      int uu  = idx & 15;
      float zi = zb[row * 64 + uu];
      float zf = zb[row * 64 + 16 + uu];
      float zg = zb[row * 64 + 32 + uu];
      float zo = zb[row * 64 + 48 + uu];
      float c  = sigmoidf(zf) * cst[idx] + sigmoidf(zi) * tanhf(zg);
      float h  = sigmoidf(zo) * tanhf(c);
      cst[idx] = c;
      hout[row * UU + u0 + uu] = to_bf16(h);
      if (t == TT - 1) hlast[row * UU + u0 + uu] = h;
    }

    // ---- device-wide step barrier ----
    __threadfence();
    __syncthreads();
    if (tid == 0) {
      atomicAdd(syncc, 1u);
      const unsigned int tgt = (unsigned int)NBLK * (unsigned int)(t + 1);
      while (__hip_atomic_load(syncc, __ATOMIC_RELAXED,
                               __HIP_MEMORY_SCOPE_AGENT) < tgt)
        __builtin_amdgcn_s_sleep(1);
    }
    __syncthreads();
  }
}

// ---------------------------------------------------------------------------
// LN2: out = LayerNorm(xn + h_last[b])   (residual broadcast over T)
// ---------------------------------------------------------------------------
__global__ __launch_bounds__(256) void ln_out_kernel(
    const float* __restrict__ xn, const float* __restrict__ hlast,
    const float* __restrict__ gamma, const float* __restrict__ beta,
    float* __restrict__ out) {
  const int row = blockIdx.x;               // b*T + t
  const int bb  = row >> 9;                 // /T
  const int tid = threadIdx.x;
  const size_t base = (size_t)row * DD + tid * 4;
  float4 v = *(const float4*)(xn + base);
  float4 h = *(const float4*)(hlast + (size_t)bb * DD + tid * 4);
  v.x += h.x; v.y += h.y; v.z += h.z; v.w += h.w;
  float s  = v.x + v.y + v.z + v.w;
  float ss = v.x * v.x + v.y * v.y + v.z * v.z + v.w * v.w;
#pragma unroll
  for (int o = 16; o > 0; o >>= 1) {
    s  += __shfl_down(s,  o, 32);
    ss += __shfl_down(ss, o, 32);
  }
  __shared__ float red[16];
  __shared__ float stat[2];
  const int lane = tid & 31, w = tid >> 5;
  if (lane == 0) { red[w] = s; red[8 + w] = ss; }
  __syncthreads();
  if (tid == 0) {
    float a = 0.f, q = 0.f;
#pragma unroll
    for (int i = 0; i < 8; ++i) { a += red[i]; q += red[8 + i]; }
    float mean = a * (1.0f / (float)DD);
    float var  = q * (1.0f / (float)DD) - mean * mean;
    stat[0] = mean;
    stat[1] = rsqrtf(var + 1e-3f);
  }
  __syncthreads();
  const float mean = stat[0], rstd = stat[1];
  float4 g = *(const float4*)(gamma + tid * 4);
  float4 b = *(const float4*)(beta + tid * 4);
  float4 y;
  y.x = (v.x - mean) * rstd * g.x + b.x;
  y.y = (v.y - mean) * rstd * g.y + b.y;
  y.z = (v.z - mean) * rstd * g.z + b.z;
  y.w = (v.w - mean) * rstd * g.w + b.w;
  *(float4*)(out + base) = y;
}

// ---------------------------------------------------------------------------
// Workspace layout (bytes)
// ---------------------------------------------------------------------------
#define OFF_XN    0u                         // fp32 xn   : 64 MB
#define OFF_XNB   67108864u                  // bf16 xn   : 32 MB
#define OFF_WB    100663296u                 // bf16 W    : 8 MB
#define OFF_RB    109051904u                 // bf16 R    : 8 MB
#define OFF_H     117440512u                 // bf16 h x2 : 128 KB
#define OFF_HLAST 117571584u                 // fp32 h_last : 128 KB
#define OFF_SYNC  117702656u                 // sync counter

extern "C" void kernel_launch(void* const* d_in, const int* in_sizes, int n_in,
                              void* d_out, int out_size, void* d_ws,
                              size_t ws_size, hipStream_t stream) {
  const float* x    = (const float*)d_in[0];
  const float* g1   = (const float*)d_in[1];
  const float* b1   = (const float*)d_in[2];
  const float* W    = (const float*)d_in[3];
  const float* R    = (const float*)d_in[4];
  const float* bias = (const float*)d_in[5];
  const float* g2   = (const float*)d_in[6];
  const float* b2   = (const float*)d_in[7];

  char* ws = (char*)d_ws;
  float*        xn    = (float*)(ws + OFF_XN);
  __bf16*       xnb   = (__bf16*)(ws + OFF_XNB);
  __bf16*       Wb    = (__bf16*)(ws + OFF_WB);
  __bf16*       Rb    = (__bf16*)(ws + OFF_RB);
  __bf16*       hbuf  = (__bf16*)(ws + OFF_H);
  float*        hlast = (float*)(ws + OFF_HLAST);
  unsigned int* syncc = (unsigned int*)(ws + OFF_SYNC);

  // zero h0, h_last, sync counter every call (deterministic, capture-safe)
  hipMemsetAsync(ws + OFF_H, 0, 131072u + 131072u + 256u, stream);

  convert_weights_kernel<<<(DD * Z4 + 255) / 256, 256, 0, stream>>>(W, R, Wb, Rb);
  ln_in_kernel<<<BB * TT, 256, 0, stream>>>(x, g1, b1, xn, xnb);

  static bool attr_set = false;  // attribute set is idempotent & not a stream op
  (void)attr_set;
  hipFuncSetAttribute((const void*)lstm_scan_kernel,
                      hipFuncAttributeMaxDynamicSharedMemorySize, LDS_BYTES);
  lstm_scan_kernel<<<NBLK, 256, LDS_BYTES, stream>>>(xnb, Wb, Rb, bias, hbuf,
                                                     hlast, syncc);

  ln_out_kernel<<<BB * TT, 256, 0, stream>>>(xn, hlast, g2, b2, (float*)d_out);
}
